// CAM_2920577761836
// MI455X (gfx1250) — matrix operand order repaired and checked by reading, not compiled, not run on hardware
//
#include <hip/hip_runtime.h>

#define CC 256
#define NN 16384

typedef __attribute__((ext_vector_type(16))) __bf16 v16bf;
typedef __attribute__((ext_vector_type(8)))  __bf16 v8bf;
typedef __attribute__((ext_vector_type(8)))  float  v8f;

__device__ __forceinline__ unsigned short f2bf(float f) {
    unsigned int u = __float_as_uint(f);
    u += 0x7FFFu + ((u >> 16) & 1u);            // round-to-nearest-even
    return (unsigned short)(u >> 16);
}

__device__ __forceinline__ unsigned int pack2bf(float f0, float f1) {
    unsigned int u0 = __float_as_uint(f0);
    unsigned int u1 = __float_as_uint(f1);
    u0 += 0x7FFFu + ((u0 >> 16) & 1u);
    u1 += 0x7FFFu + ((u1 >> 16) & 1u);
    return (u0 >> 16) | (u1 & 0xFFFF0000u);     // [bf(f1) : bf(f0)]
}

__device__ __forceinline__ v16bf ld_frag16(const unsigned short* p0,
                                           const unsigned short* p1) {
    v8bf lo = *reinterpret_cast<const v8bf*>(p0);
    v8bf hi = *reinterpret_cast<const v8bf*>(p1);
    return __builtin_shufflevector(lo, hi, 0,1,2,3,4,5,6,7,8,9,10,11,12,13,14,15);
}

// ---- Stage 1: energy[b] = X[b] * X[b]^T, symmetric (upper-triangular blocks) --
__global__ __launch_bounds__(512) void cam_gram(const float* __restrict__ x,
                                                float* __restrict__ energy) {
    // map blockIdx.x (0..9) -> upper-triangular 64x64 tile pair (it <= jt), 4x4 grid
    int p = blockIdx.x, it = 0;
    while (p >= 4 - it) { p -= (4 - it); ++it; }
    const int jt = it + p;

    const int b    = blockIdx.z;
    const int i0   = it * 64;
    const int j0   = jt * 64;
    const int t    = threadIdx.x;
    const int lane = t & 31;
    const int wave = t >> 5;                 // 0..15
    const int wi   = (wave >> 2) * 16;
    const int wj   = (wave & 3) * 16;

    __shared__ unsigned short ldsA[64][72];  // 144B row stride (16B aligned)
    __shared__ unsigned short ldsB[64][72];

    const float* xb = x + (size_t)b * CC * NN;

    v8f acc = {};

    // staging map: thread -> (row, 8 consecutive cols)
    const int srow = t >> 3;                 // 0..63
    const int scol = (t & 7) * 8;            // 0..56

    // fragment lane maps (ISA 7.12.2)
    const int m  = wi + (lane & 15);
    const int n  = wj + (lane & 15);
    const int hb = (lane >> 4) * 8;
    const int kb = (lane >> 4) * 16;

    for (int kc = 0; kc < NN; kc += 64) {
        const float4 a0 = *(const float4*)&xb[(size_t)(i0 + srow) * NN + kc + scol];
        const float4 a1 = *(const float4*)&xb[(size_t)(i0 + srow) * NN + kc + scol + 4];
        const float4 b0 = *(const float4*)&xb[(size_t)(j0 + srow) * NN + kc + scol];
        const float4 b1 = *(const float4*)&xb[(size_t)(j0 + srow) * NN + kc + scol + 4];
        if (kc + 64 < NN) {                  // prefetch next K strip (global_prefetch_b8)
            __builtin_prefetch(&xb[(size_t)(i0 + srow) * NN + kc + 64 + scol], 0, 1);
            __builtin_prefetch(&xb[(size_t)(j0 + srow) * NN + kc + 64 + scol], 0, 1);
        }
        uint4 pa = { pack2bf(a0.x, a0.y), pack2bf(a0.z, a0.w),
                     pack2bf(a1.x, a1.y), pack2bf(a1.z, a1.w) };
        uint4 pb = { pack2bf(b0.x, b0.y), pack2bf(b0.z, b0.w),
                     pack2bf(b1.x, b1.y), pack2bf(b1.z, b1.w) };
        __syncthreads();
        *reinterpret_cast<uint4*>(&ldsA[srow][scol]) = pa;
        *reinterpret_cast<uint4*>(&ldsB[srow][scol]) = pb;
        __syncthreads();
#pragma unroll
        for (int kk = 0; kk < 64; kk += 32) {
            v16bf a  = ld_frag16(&ldsA[m][kk + hb], &ldsA[m][kk + 16 + hb]);
            v16bf bf = ld_frag16(&ldsB[n][kk + hb], &ldsB[n][kk + 16 + hb]);
            acc = __builtin_amdgcn_wmma_f32_16x16x32_bf16(false, a, false, bf,
                                                          (short)0, acc, false, false);
        }
    }

    float* eb = energy + (size_t)b * CC * CC;
    const int col   = j0 + wj + (lane & 15);
    const int rbase = i0 + wi + (lane >> 4) * 8;
#pragma unroll
    for (int r = 0; r < 8; ++r)
        eb[(size_t)(rbase + r) * CC + col] = acc[r];
    if (it != jt) {                          // mirror to lower triangle
#pragma unroll
        for (int r = 0; r < 8; ++r)
            eb[(size_t)col * CC + (rbase + r)] = acc[r];
    }
}

// ---- Stage 2: attention = softmax(rowmax - energy) = exp(rowmin - e) / sum ----
__global__ __launch_bounds__(256) void cam_softmax(const float* __restrict__ energy,
                                                   unsigned short* __restrict__ attB) {
    const int row = blockIdx.x;              // b*256 + i
    const int t   = threadIdx.x;
    __shared__ float red[256];

    float e = energy[(size_t)row * CC + t];

    red[t] = e; __syncthreads();             // row min
    for (int s = 128; s > 0; s >>= 1) {
        if (t < s) red[t] = fminf(red[t], red[t + s]);
        __syncthreads();
    }
    float mn = red[0]; __syncthreads();

    float pr = __expf(mn - e);
    red[t] = pr; __syncthreads();            // row sum
    for (int s = 128; s > 0; s >>= 1) {
        if (t < s) red[t] += red[t + s];
        __syncthreads();
    }
    float inv = 1.0f / red[0];

    attB[(size_t)row * CC + t] = f2bf(pr * inv);
}

// ---- Stage 3: out = gamma * (att @ X) + x -------------------------------------
__global__ __launch_bounds__(512) void cam_av(const float* __restrict__ x,
                                              const unsigned short* __restrict__ attB,
                                              const float* __restrict__ gamma,
                                              float* __restrict__ out) {
    const int b    = blockIdx.z;
    const int i0   = blockIdx.y * 64;        // M tile (channels)
    const int n0   = blockIdx.x * 256;       // N tile (spatial)
    const int t    = threadIdx.x;
    const int lane = t & 31;
    const int wave = t >> 5;
    const int wi   = (wave >> 2) * 16;
    const int wn   = (wave & 3) * 64;

    __shared__ unsigned short ldsA[64][40];  // att tile [i][j], 80B stride
    __shared__ unsigned short ldsX[256][40]; // X tile transposed [n][k]

    const float* xb          = x    + (size_t)b * CC * NN;
    const unsigned short* ab = attB + (size_t)b * CC * CC;

    v8f acc[4] = {{}, {}, {}, {}};

    const int arow = t >> 3;                 // A staging: 4 bf16 per thread
    const int acol = (t & 7) * 4;

    const int m  = wi + (lane & 15);
    const int hb = (lane >> 4) * 8;
    const int kb = (lane >> 4) * 16;

    for (int j0 = 0; j0 < CC; j0 += 32) {
        __syncthreads();
        // stage attention 64x32 (already bf16): one 8B copy per thread
        *reinterpret_cast<uint2*>(&ldsA[arow][acol]) =
            *reinterpret_cast<const uint2*>(&ab[(size_t)(i0 + arow) * CC + j0 + acol]);
        // stage X 32x256 -> transposed bf16: float4 coalesced reads along n
#pragma unroll
        for (int r = 0; r < 4; ++r) {
            int q  = t + r * 512;            // 0..2047 quads
            int k  = q >> 6;                 // 0..31
            int nq = (q & 63) * 4;           // 0..252
            const float4 v = *(const float4*)&xb[(size_t)(j0 + k) * NN + n0 + nq];
            ldsX[nq + 0][k] = f2bf(v.x);
            ldsX[nq + 1][k] = f2bf(v.y);
            ldsX[nq + 2][k] = f2bf(v.z);
            ldsX[nq + 3][k] = f2bf(v.w);
        }
        __syncthreads();

        v16bf a = ld_frag16(&ldsA[m][hb], &ldsA[m][16 + hb]);
#pragma unroll
        for (int s = 0; s < 4; ++s) {
            int n = wn + s * 16 + (lane & 15);
            v16bf bf = ld_frag16(&ldsX[n][hb], &ldsX[n][16 + hb]);
            acc[s] = __builtin_amdgcn_wmma_f32_16x16x32_bf16(false, a, false, bf,
                                                             (short)0, acc[s], false, false);
        }
    }

    const float g     = gamma[0];
    const int   rbase = i0 + wi + (lane >> 4) * 8;
#pragma unroll
    for (int s = 0; s < 4; ++s) {
        int col = n0 + wn + s * 16 + (lane & 15);
#pragma unroll
        for (int r = 0; r < 8; ++r) {
            size_t idx = (size_t)b * CC * NN + (size_t)(rbase + r) * NN + col;
            out[idx] = g * acc[s][r] + x[idx];
        }
    }
}

extern "C" void kernel_launch(void* const* d_in, const int* in_sizes, int n_in,
                              void* d_out, int out_size, void* d_ws, size_t ws_size,
                              hipStream_t stream) {
    const float* x     = (const float*)d_in[0];
    const float* gamma = (const float*)d_in[1];
    float*       out   = (float*)d_out;

    float*          energy = (float*)d_ws;                                   // 4 MB
    unsigned short* attB   = (unsigned short*)((char*)d_ws +
                              (size_t)16 * CC * CC * sizeof(float));         // +2 MB

    cam_gram<<<dim3(10, 1, 16), 512, 0, stream>>>(x, energy);      // 10 tri-tiles/batch

    cam_softmax<<<dim3(16 * CC), 256, 0, stream>>>(energy, attB);

    cam_av<<<dim3(NN / 256, CC / 64, 16), 512, 0, stream>>>(x, attB, gamma, out);
}
